// GATModel_34265249087668
// MI455X (gfx1250) — matrix operand order repaired and checked by reading, not compiled, not run on hardware
//
#include <hip/hip_runtime.h>
#include <hip/hip_bf16.h>

// ---------------------------------------------------------------------------
// GATv2 2-layer GNN for MI455X (gfx1250, wave32).
// GEMMs via v_wmma_f32_16x16x32_bf16 on pre-converted bf16 activations/weights
// (inner loop = pure global_load_b128 + wmma); edge softmax/aggregation via
// L2-resident bf16 gathers + fp32 atomics; fused layernorm/ELU node epilogues.
// ---------------------------------------------------------------------------

typedef __attribute__((ext_vector_type(16))) __bf16 v16bf;
typedef __attribute__((ext_vector_type(8)))  float  v8f;

union FragBF { v16bf v; unsigned u[8]; uint4 q[2]; };
union FragF  { v8f   v; float     f[8]; };

__device__ __forceinline__ unsigned f2bf(float f) {
    unsigned x = __float_as_uint(f);
    return (x + 0x7FFFu + ((x >> 16) & 1u)) >> 16;   // RNE
}
__device__ __forceinline__ unsigned pack2bf(float a, float b) {
    return f2bf(a) | (f2bf(b) << 16);
}
__device__ __forceinline__ float bflo(unsigned u) { return __uint_as_float(u << 16); }
__device__ __forceinline__ float bfhi(unsigned u) { return __uint_as_float(u & 0xFFFF0000u); }

// A fragment (16x32 bf16) from bf16 rows [row][k], ld = 128.
// Layout: lanes 0-15 hold K=0..15, lanes 16-31 K=16..31 (matches load_b_bf16).
__device__ __forceinline__ FragBF load_a_bf16(const unsigned short* __restrict__ base,
                                              int row, int k0) {
    FragBF a;
    int half = (threadIdx.x >> 4) & 1;
    const unsigned short* p = base + (size_t)row * 128 + k0 + half * 16;
    a.q[0] = *(const uint4*)p;
    a.q[1] = *(const uint4*)(p + 8);
    return a;
}

// B fragment (32x16 bf16) from pre-transposed bf16 weights Wt[n][k].
// Contiguous-k layout (same as A, not ISA table order): lanes 0-15 hold K=0..15 (VGPR j = K 2j,2j+1), lanes 16-31 K=16..31.
__device__ __forceinline__ FragBF load_b_bf16(const unsigned short* __restrict__ wt,
                                              int n_base, int k0) {
    FragBF b;
    int lane = threadIdx.x & 31;
    int col  = n_base + (lane & 15);
    int half = lane >> 4;
    const unsigned short* p = wt + (size_t)col * 128 + k0 + half * 16;
    b.q[0] = *(const uint4*)p;
    b.q[1] = *(const uint4*)(p + 8);
    return b;
}

// ---------------------------------------------------------------------------
// 0a) weight prep: fp32 [K][Nout] -> bf16 [Nout][K=128] (transposed)
// ---------------------------------------------------------------------------
__global__ void gat_prep_weights(const float* Wl1, const float* Wr1, const float* res1W,
                                 const float* Wl2, const float* Wr2, const float* res2W,
                                 const float* skW, unsigned short* wdst) {
    const float* src; int nout; size_t doff;
    switch (blockIdx.y) {
        case 0: src = Wl1;   nout = 128; doff = 0;            break;
        case 1: src = Wr1;   nout = 128; doff = 16384;        break;
        case 2: src = res1W; nout = 128; doff = 32768;        break;
        case 3: src = Wl2;   nout = 16;  doff = 49152;        break;
        case 4: src = Wr2;   nout = 16;  doff = 49152 + 2048; break;
        case 5: src = res2W; nout = 16;  doff = 49152 + 4096; break;
        default: src = skW;  nout = 16;  doff = 49152 + 6144; break;
    }
    int t = blockIdx.x * blockDim.x + threadIdx.x;
    if (t >= 128 * nout) return;
    int n = t / 128, k = t % 128;
    wdst[doff + (size_t)n * 128 + k] = (unsigned short)f2bf(src[(size_t)k * nout + n]);
}

// ---------------------------------------------------------------------------
// 0b) activation prep: x fp32 [N][128] -> bf16 [N][128]
// ---------------------------------------------------------------------------
__global__ void gat_prep_x(const float* __restrict__ x, unsigned short* __restrict__ xbf,
                           long long total4) {
    long long t = (long long)blockIdx.x * blockDim.x + threadIdx.x;
    if (t >= total4) return;
    float4 v = *(const float4*)(x + t * 4);
    uint2 o;
    o.x = pack2bf(v.x, v.y);
    o.y = pack2bf(v.z, v.w);
    *(uint2*)(xbf + t * 4) = o;
}

// ---------------------------------------------------------------------------
// 1) layer-1 node transforms: xl1 = x@Wl1+bl1, xr1 = x@Wr1+br1, res1 = x@res1_W+res1_b
//    one wave per 16-row tile, blockIdx.y selects the matrix; bf16 outputs.
// ---------------------------------------------------------------------------
__global__ void gat_transform1(const unsigned short* __restrict__ xbf,
                               const unsigned short* __restrict__ wts,
                               const float* bl1, const float* br1, const float* res1b,
                               unsigned short* xl1, unsigned short* xr1, unsigned short* res1,
                               int ntiles) {
    int rt = blockIdx.x * (blockDim.x >> 5) + (threadIdx.x >> 5);
    if (rt >= ntiles) return;                       // wave-uniform
    int m = blockIdx.y;
    const unsigned short* wt = wts + (size_t)m * 16384;
    const float* bias   = (m == 0) ? bl1 : (m == 1) ? br1 : res1b;
    unsigned short* outp = (m == 0) ? xl1 : (m == 1) ? xr1 : res1;

    __builtin_prefetch(wt, 0, 3);                   // global_prefetch_b8

    int lane = threadIdx.x & 31;
    int r = lane & 15, half = lane >> 4;
    int row_base = rt * 16;

    FragF acc[8];
#pragma unroll
    for (int t = 0; t < 8; ++t)
#pragma unroll
        for (int g = 0; g < 8; ++g) acc[t].f[g] = 0.f;

#pragma unroll
    for (int ks = 0; ks < 4; ++ks) {
        int k0 = ks * 32;
        FragBF a = load_a_bf16(xbf, row_base + r, k0);
#pragma unroll
        for (int t = 0; t < 8; ++t) {
            FragBF b = load_b_bf16(wt, t * 16, k0);
            acc[t].v = __builtin_amdgcn_wmma_f32_16x16x32_bf16(
                false, a.v, false, b.v, (short)0, acc[t].v, false, false);
        }
    }
#pragma unroll
    for (int t = 0; t < 8; ++t) {
        int n = t * 16 + r;
        float bv = bias[n];
#pragma unroll
        for (int g = 0; g < 8; ++g) {
            int rowo = row_base + g + 8 * half;     // C/D layout: VGPR g -> row g(+8)
            outp[(size_t)rowo * 128 + n] = (unsigned short)f2bf(acc[t].f[g] + bv);
        }
    }
}

// ---------------------------------------------------------------------------
// 2) layer-1 edge pass: wave per edge. lane -> (head h, quarter q).
//    alpha = att1[h] . leakyrelu(xl1[src]+xr1[dst]); w = exp(alpha)
//    atomics: denom1[dst][h] += w ; out1[dst][c] += xl1[src][c] * w
// ---------------------------------------------------------------------------
__global__ void gat_edge1(const int* __restrict__ ei, int E, int Nn,
                          const unsigned short* __restrict__ xl1,
                          const unsigned short* __restrict__ xr1,
                          const float* __restrict__ att1,
                          float* __restrict__ out1, float* __restrict__ denom1) {
    long long e = (long long)blockIdx.x * (blockDim.x >> 5) + (threadIdx.x >> 5);
    if (e >= (long long)E + Nn) return;
    int src, dst;
    if (e < E) { src = ei[e]; dst = ei[(size_t)E + e]; }
    else       { src = dst = (int)(e - E); }

    int lane = threadIdx.x & 31;
    int h = lane >> 2, q = lane & 3;
    int c = h * 16 + q * 4;

    const unsigned* pl = (const unsigned*)(xl1 + (size_t)src * 128 + c);
    const unsigned* pr = (const unsigned*)(xr1 + (size_t)dst * 128 + c);
    unsigned l0 = pl[0], l1 = pl[1], r0 = pr[0], r1 = pr[1];
    float xls[4] = { bflo(l0), bfhi(l0), bflo(l1), bfhi(l1) };
    float xrs[4] = { bflo(r0), bfhi(r0), bflo(r1), bfhi(r1) };
    float4 av = *(const float4*)(att1 + c);
    float aa[4] = { av.x, av.y, av.z, av.w };

    float partial = 0.f;
#pragma unroll
    for (int j = 0; j < 4; ++j) {
        float mm = xls[j] + xrs[j];
        float ee = mm > 0.f ? mm : 0.2f * mm;       // leaky_relu(0.2)
        partial += ee * aa[j];
    }
    partial += __shfl_xor(partial, 1, 32);
    partial += __shfl_xor(partial, 2, 32);
    float w = __expf(partial);                      // max-free softmax (logits O(0.1))

    if (q == 0) atomicAdd(denom1 + (size_t)dst * 8 + h, w);
    float* po = out1 + (size_t)dst * 128 + c;
#pragma unroll
    for (int j = 0; j < 4; ++j) atomicAdd(po + j, xls[j] * w);
}

// ---------------------------------------------------------------------------
// 3) layer-1 node finalize: normalize, +bias1, LayerNorm(128), +res1, ELU.
//    Writes h as bf16 in place over the (dead) res1 region.
// ---------------------------------------------------------------------------
__global__ void gat_finalize1(const float* __restrict__ out1,
                              const float* __restrict__ denom1, const float* __restrict__ bias1,
                              const float* __restrict__ ln1g, const float* __restrict__ ln1b,
                              unsigned short* res1_hbf, int Nn) {
    int n = blockIdx.x * (blockDim.x >> 5) + (threadIdx.x >> 5);
    if (n >= Nn) return;
    int lane = threadIdx.x & 31;
    int c = lane * 4, h = lane >> 2;

    float4 o = *(const float4*)(out1 + (size_t)n * 128 + c);
    float inv = 1.f / (denom1[(size_t)n * 8 + h] + 1e-16f);
    float v[4] = { o.x * inv + bias1[c],     o.y * inv + bias1[c + 1],
                   o.z * inv + bias1[c + 2], o.w * inv + bias1[c + 3] };

    float s1 = v[0] + v[1] + v[2] + v[3];
    float s2 = v[0]*v[0] + v[1]*v[1] + v[2]*v[2] + v[3]*v[3];
#pragma unroll
    for (int m = 16; m >= 1; m >>= 1) {
        s1 += __shfl_xor(s1, m, 32);
        s2 += __shfl_xor(s2, m, 32);
    }
    float mu   = s1 * (1.f / 128.f);
    float var  = s2 * (1.f / 128.f) - mu * mu;
    float rstd = rsqrtf(var + 1e-5f);

    unsigned* prw = (unsigned*)(res1_hbf + (size_t)n * 128 + c);
    unsigned q0 = prw[0], q1 = prw[1];
    float rs[4] = { bflo(q0), bfhi(q0), bflo(q1), bfhi(q1) };
#pragma unroll
    for (int j = 0; j < 4; ++j) {
        float t = (v[j] - mu) * rstd * ln1g[c + j] + ln1b[c + j] + rs[j];
        v[j] = t > 0.f ? t : (__expf(t) - 1.f);     // ELU
    }
    prw[0] = pack2bf(v[0], v[1]);                   // h (bf16) overwrites res1
    prw[1] = pack2bf(v[2], v[3]);
}

// ---------------------------------------------------------------------------
// 4) layer-2 node transforms: hl2/hr2/r2/sk = h @ {Wl2,Wr2,res2_W,skip_W} + bias
//    one wave per 16-row tile, all 4 matrices accumulated in-wave (n-dim = 16).
// ---------------------------------------------------------------------------
__global__ void gat_transform2(const unsigned short* __restrict__ hbf,
                               const unsigned short* __restrict__ wts2,
                               const float* bl2, const float* br2, const float* res2b, const float* skb,
                               float* hl2, float* hr2, float* r2, float* sk, int ntiles) {
    int rt = blockIdx.x * (blockDim.x >> 5) + (threadIdx.x >> 5);
    if (rt >= ntiles) return;                       // wave-uniform
    int lane = threadIdx.x & 31;
    int r = lane & 15, half = lane >> 4;
    int row_base = rt * 16;

    FragF acc[4];
#pragma unroll
    for (int t = 0; t < 4; ++t)
#pragma unroll
        for (int g = 0; g < 8; ++g) acc[t].f[g] = 0.f;

#pragma unroll
    for (int ks = 0; ks < 4; ++ks) {
        int k0 = ks * 32;
        FragBF a = load_a_bf16(hbf, row_base + r, k0);
#pragma unroll
        for (int mi = 0; mi < 4; ++mi) {
            FragBF b = load_b_bf16(wts2 + (size_t)mi * 2048, 0, k0);
            acc[mi].v = __builtin_amdgcn_wmma_f32_16x16x32_bf16(
                false, a.v, false, b.v, (short)0, acc[mi].v, false, false);
        }
    }
    const float* biases[4] = { bl2, br2, res2b, skb };
    float* outs[4] = { hl2, hr2, r2, sk };
#pragma unroll
    for (int mi = 0; mi < 4; ++mi) {
        float bv = biases[mi][r];
#pragma unroll
        for (int g = 0; g < 8; ++g) {
            int rowo = row_base + g + 8 * half;
            outs[mi][(size_t)rowo * 16 + r] = acc[mi].f[g] + bv;
        }
    }
}

// ---------------------------------------------------------------------------
// 5) layer-2 edge pass (H=1, C=16): 4 lanes per edge.
// ---------------------------------------------------------------------------
__global__ void gat_edge2(const int* __restrict__ ei, int E, int Nn,
                          const float* __restrict__ hl2, const float* __restrict__ hr2,
                          const float* __restrict__ att2,
                          float* __restrict__ out2, float* __restrict__ denom2) {
    long long t = (long long)blockIdx.x * blockDim.x + threadIdx.x;
    long long e = t >> 2;
    if (e >= (long long)E + Nn) return;
    int q = (int)(t & 3);
    int src, dst;
    if (e < E) { src = ei[e]; dst = ei[(size_t)E + e]; }
    else       { src = dst = (int)(e - E); }
    int c = q * 4;

    float4 lv = *(const float4*)(hl2 + (size_t)src * 16 + c);
    float4 rv = *(const float4*)(hr2 + (size_t)dst * 16 + c);
    float4 av = *(const float4*)(att2 + c);
    float xl[4] = { lv.x, lv.y, lv.z, lv.w };
    float xr[4] = { rv.x, rv.y, rv.z, rv.w };
    float aa[4] = { av.x, av.y, av.z, av.w };

    float partial = 0.f;
#pragma unroll
    for (int j = 0; j < 4; ++j) {
        float mm = xl[j] + xr[j];
        float ee = mm > 0.f ? mm : 0.2f * mm;
        partial += ee * aa[j];
    }
    partial += __shfl_xor(partial, 1, 32);
    partial += __shfl_xor(partial, 2, 32);
    float w = __expf(partial);

    if (q == 0) atomicAdd(denom2 + dst, w);
    float* po = out2 + (size_t)dst * 16 + c;
#pragma unroll
    for (int j = 0; j < 4; ++j) atomicAdd(po + j, xl[j] * w);
}

// ---------------------------------------------------------------------------
// 6) layer-2 node finalize + output projection (16 -> 64): one 64-thread block/node.
// ---------------------------------------------------------------------------
__global__ void gat_finalize2(const float* __restrict__ out2, const float* __restrict__ denom2,
                              const float* __restrict__ bias2,
                              const float* __restrict__ ln2g, const float* __restrict__ ln2b,
                              const float* __restrict__ r2, const float* __restrict__ sk,
                              const float* __restrict__ outW, const float* __restrict__ outb,
                              float* __restrict__ y, int Nn) {
    int n = blockIdx.x;
    if (n >= Nn) return;
    __shared__ float sh[16];
    __shared__ float smu, srstd;
    int t = threadIdx.x;
    if (t < 16)
        sh[t] = out2[(size_t)n * 16 + t] / (denom2[n] + 1e-16f) + bias2[t];
    __syncthreads();
    if (t == 0) {
        float s1 = 0.f, s2 = 0.f;
        for (int i = 0; i < 16; ++i) { s1 += sh[i]; s2 += sh[i] * sh[i]; }
        float mu = s1 * (1.f / 16.f);
        smu = mu;
        srstd = rsqrtf(s2 * (1.f / 16.f) - mu * mu + 1e-5f);
    }
    __syncthreads();
    if (t < 16) {
        float v = (sh[t] - smu) * srstd * ln2g[t] + ln2b[t] + r2[(size_t)n * 16 + t];
        v = v > 0.f ? v : (__expf(v) - 1.f);        // ELU
        sh[t] = v + sk[(size_t)n * 16 + t];
    }
    __syncthreads();
    float acc = outb[t];
#pragma unroll
    for (int cix = 0; cix < 16; ++cix) acc += sh[cix] * outW[cix * 64 + t];
    y[(size_t)n * 64 + t] = acc;
}

// ---------------------------------------------------------------------------
extern "C" void kernel_launch(void* const* d_in, const int* in_sizes, int n_in,
                              void* d_out, int out_size, void* d_ws, size_t ws_size,
                              hipStream_t stream) {
    (void)n_in; (void)out_size; (void)ws_size;
    const float* x     = (const float*)d_in[0];
    const int*   ei    = (const int*)d_in[1];
    const float* Wl1   = (const float*)d_in[2];  const float* bl1   = (const float*)d_in[3];
    const float* Wr1   = (const float*)d_in[4];  const float* br1   = (const float*)d_in[5];
    const float* att1  = (const float*)d_in[6];  const float* bias1 = (const float*)d_in[7];
    const float* Wl2   = (const float*)d_in[8];  const float* bl2   = (const float*)d_in[9];
    const float* Wr2   = (const float*)d_in[10]; const float* br2   = (const float*)d_in[11];
    const float* att2  = (const float*)d_in[12]; const float* bias2 = (const float*)d_in[13];
    const float* ln1g  = (const float*)d_in[14]; const float* ln1b  = (const float*)d_in[15];
    const float* ln2g  = (const float*)d_in[16]; const float* ln2b  = (const float*)d_in[17];
    const float* res1W = (const float*)d_in[18]; const float* res1b = (const float*)d_in[19];
    const float* res2W = (const float*)d_in[20]; const float* res2b = (const float*)d_in[21];
    const float* skW   = (const float*)d_in[22]; const float* skb   = (const float*)d_in[23];
    const float* outW  = (const float*)d_in[24]; const float* outb  = (const float*)d_in[25];
    float* y = (float*)d_out;

    int Nn = in_sizes[0] / 128;      // 100000
    int E  = in_sizes[1] / 2;        // 1600000
    int totalE = E + Nn;
    int ntiles = Nn / 16;            // 6250

    // workspace layout (~131.3 MB, region-aliased):
    //  A [0,       szA)  : xl1 bf16          -> later hl2/hr2/r2/sk f32
    //  B [szA,    2szA)  : xr1 bf16          -> later out2 + denom2 f32
    //  C [2szA,   3szA)  : res1 bf16         -> overwritten in place by h bf16
    //  D [3szA, 3szA+szD): xbf bf16 (prep)   -> then zeroed, out1 f32 accumulator
    //  E [.. +szE)       : denom1 f32
    //  F [..]            : bf16 weights (~112 KB)
    char* ws = (char*)d_ws;
    size_t szA = (size_t)Nn * 128 * 2;
    size_t szD = (size_t)Nn * 128 * 4;
    size_t szE = (size_t)Nn * 8 * 4;
    unsigned short* xl1  = (unsigned short*)(ws);
    unsigned short* xr1  = (unsigned short*)(ws + szA);
    unsigned short* res1 = (unsigned short*)(ws + 2 * szA);   // becomes h (bf16)
    float* out1   = (float*)(ws + 3 * szA);
    unsigned short* xbf = (unsigned short*)(ws + 3 * szA);    // aliases out1 (pre-memset)
    float* denom1 = (float*)(ws + 3 * szA + szD);
    unsigned short* wts = (unsigned short*)(ws + 3 * szA + szD + szE);
    size_t sz16 = (size_t)Nn * 16 * 4;
    float* hl2 = (float*)(ws);
    float* hr2 = (float*)(ws + sz16);
    float* r2  = (float*)(ws + 2 * sz16);
    float* sk  = (float*)(ws + 3 * sz16);
    float* out2   = (float*)(ws + szA);
    float* denom2 = (float*)(ws + szA + sz16);

    // 0) weights + activations -> bf16
    gat_prep_weights<<<dim3(64, 7), 256, 0, stream>>>(Wl1, Wr1, res1W, Wl2, Wr2, res2W, skW, wts);
    long long total4 = (long long)Nn * 32;
    gat_prep_x<<<dim3((unsigned)((total4 + 255) / 256)), 256, 0, stream>>>(x, xbf, total4);
    // 1) layer-1 transforms (WMMA, pure bf16 loads)
    gat_transform1<<<dim3((ntiles + 7) / 8, 3), 256, 0, stream>>>(
        xbf, wts, bl1, br1, res1b, xl1, xr1, res1, ntiles);
    // zero layer-1 accumulators (xbf is dead now; out1 + denom1 contiguous)
    hipMemsetAsync(out1, 0, szD + szE, stream);
    // 2) layer-1 edges
    gat_edge1<<<dim3((totalE + 7) / 8), 256, 0, stream>>>(ei, E, Nn, xl1, xr1, att1, out1, denom1);
    // 3) layer-1 node finalize -> h (bf16, in res1 region)
    gat_finalize1<<<dim3((Nn + 7) / 8), 256, 0, stream>>>(out1, denom1, bias1, ln1g, ln1b, res1, Nn);
    // zero layer-2 accumulators (out2 + denom2, contiguous)
    hipMemsetAsync(out2, 0, sz16 + (size_t)Nn * 4, stream);
    // 4) layer-2 transforms (WMMA)
    gat_transform2<<<dim3((ntiles + 7) / 8), 256, 0, stream>>>(
        res1, wts + 3 * 16384, bl2, br2, res2b, skb, hl2, hr2, r2, sk, ntiles);
    // 5) layer-2 edges
    gat_edge2<<<dim3((totalE * 4 + 255) / 256), 256, 0, stream>>>(ei, E, Nn, hl2, hr2, att2, out2, denom2);
    // 6) finalize + output projection
    gat_finalize2<<<dim3(Nn), 64, 0, stream>>>(out2, denom2, bias2, ln2g, ln2b, r2, sk, outW, outb, y, Nn);
}
